// RechitGNN_22892175688441
// MI455X (gfx1250) — matrix operand order repaired and checked by reading, not compiled, not run on hardware
//
#include <hip/hip_runtime.h>

#define B_    16
#define NPER  1024
#define NTOT  (B_ * NPER)
#define KNN   16
#define D_    128

typedef __attribute__((ext_vector_type(16))) __bf16 bf16x16;
typedef __attribute__((ext_vector_type(8)))  float  f32x8;

__device__ __forceinline__ f32x8 wmma_bf16(bf16x16 a, bf16x16 b, f32x8 c) {
  // D = A(16x32 bf16) * B(32x16 bf16) + C(16x16 f32)
  return __builtin_amdgcn_wmma_f32_16x16x32_bf16(
      /*neg_a=*/false, a, /*neg_b=*/false, b,
      /*c_mod=*/(short)0, c, /*reuse_a=*/false, /*reuse_b=*/false);
}

// Sorted-ascending top-16 insertion (register-resident; the 16-deep network
// MUST be unrolled so bd/bi stay in VGPRs).
#define TOPK_INSERT16(bd, bi, dval, ival)           \
  do {                                              \
    float _cd = (dval); int _ci = (ival);           \
    if (_cd < bd[15]) {                             \
      _Pragma("unroll")                             \
      for (int _k = 0; _k < 16; ++_k) {             \
        if (_cd < bd[_k]) {                         \
          float _td = bd[_k]; int _ti = bi[_k];     \
          bd[_k] = _cd; bi[_k] = _ci;               \
          _cd = _td; _ci = _ti;                     \
        }                                           \
      }                                             \
    }                                               \
  } while (0)

// A fragment: 16x32 tile, row-major source (element (m,k) = src[m*lda + k]).
// Lane m%16 holds row m; lanes 0-15 carry K 0-15, lanes 16-31 carry K 16-31.
__device__ __forceinline__ bf16x16 load_a_frag(const __bf16* src, int lda) {
  int lane = threadIdx.x & 31;
  int m = lane & 15;
  int kbase = (lane < 16) ? 0 : 16;
  bf16x16 a;
#pragma unroll
  for (int v = 0; v < 8; ++v) {
    int k = kbase + 2 * v;
    a[2 * v]     = src[m * lda + k];
    a[2 * v + 1] = src[m * lda + k + 1];
  }
  return a;
}

// B fragment: 32x16 tile, element (k,n) = src[k*ldb + n] (row-major K x N source).
// Lane n holds column n; lanes 0-15 carry K 0-15, lanes 16-31 carry K 16-31.
__device__ __forceinline__ bf16x16 load_b_frag(const __bf16* src, int ldb) {
  int lane = threadIdx.x & 31;
  int n = lane & 15;
  int kbase = (lane < 16) ? 0 : 16;
  bf16x16 b;
#pragma unroll
  for (int v = 0; v < 8; ++v) {
    int k = kbase + 2 * v;
    b[2 * v]     = src[k * ldb + n];
    b[2 * v + 1] = src[(k + 1) * ldb + n];
  }
  return b;
}

// B fragment for H^T (Gram): element (k,n) = src[n*ld + k].
__device__ __forceinline__ bf16x16 load_b_frag_T(const __bf16* src, int ld) {
  int lane = threadIdx.x & 31;
  int n = lane & 15;
  int kbase = (lane < 16) ? 0 : 16;
  bf16x16 b;
#pragma unroll
  for (int v = 0; v < 8; ++v) {
    int k = kbase + 2 * v;
    b[2 * v]     = src[n * ld + k];
    b[2 * v + 1] = src[n * ld + k + 1];
  }
  return b;
}

// ---------------- encoder: h = relu(relu(x@W1+b1)@W2+b2), 4->64->128 -------
__global__ __launch_bounds__(256) void encoder_kernel(
    const float* __restrict__ x, const float* __restrict__ w1,
    const float* __restrict__ b1, const float* __restrict__ w2,
    const float* __restrict__ b2, float* __restrict__ h) {
  int i = blockIdx.x * blockDim.x + threadIdx.x;
  if (i >= NTOT) return;
  float xi[4];
#pragma unroll
  for (int k = 0; k < 4; ++k) xi[k] = x[i * 4 + k];
  float t[64];
#pragma unroll 4
  for (int j = 0; j < 64; ++j) {
    float a = b1[j];
#pragma unroll
    for (int k = 0; k < 4; ++k) a += xi[k] * w1[k * 64 + j];
    t[j] = fmaxf(a, 0.f);
  }
#pragma unroll 2
  for (int j = 0; j < 128; ++j) {
    float a = b2[j];
#pragma unroll 8
    for (int k = 0; k < 64; ++k) a += t[k] * w2[k * 128 + j];
    h[(size_t)i * D_ + j] = fmaxf(a, 0.f);
  }
}

// ---------------- fp32 -> bf16 weight conversion ---------------------------
__global__ void cvt_bf16_kernel(const float* __restrict__ src,
                                __bf16* __restrict__ dst, int n) {
  int i = blockIdx.x * blockDim.x + threadIdx.x;
  if (i < n) dst[i] = (__bf16)src[i];
}

// ---------------- per-layer prep: bf16 features + squared norms ------------
__global__ __launch_bounds__(256) void prep_kernel(
    const float* __restrict__ hb, __bf16* __restrict__ hb16,
    float* __restrict__ norms) {
  int pt = blockIdx.x * 8 + (threadIdx.x >> 5);  // one wave per point
  int lane = threadIdx.x & 31;
  float s = 0.f;
#pragma unroll
  for (int q = 0; q < 4; ++q) {
    int c = lane + 32 * q;
    float v = hb[(size_t)pt * D_ + c];
    __bf16 bv = (__bf16)v;
    float vr = (float)bv;          // norm consistent with bf16-rounded dot
    s += vr * vr;
    hb16[(size_t)pt * D_ + c] = bv;
  }
#pragma unroll
  for (int off = 16; off >= 1; off >>= 1) s += __shfl_xor(s, off, 32);
  if (lane == 0) norms[pt] = s;
}

// ---------------- kNN: WMMA Gram + two-stage parallel top-16 ---------------
// grid = B_*(NPER/16) blocks of 256 threads (8 waves).
__global__ __launch_bounds__(256) void knn_kernel(
    const __bf16* __restrict__ hb16, const float* __restrict__ norms,
    int* __restrict__ idx_out) {
  __shared__ float dist[16 * NPER];  // 64 KB strip: 16 rows x 1024 cols
  int b  = blockIdx.x >> 6;
  int rt = blockIdx.x & 63;
  int row0 = rt * 16;
  const __bf16* base = hb16 + (size_t)b * NPER * D_;
  const float*  nrm  = norms + b * NPER;
  int wave = threadIdx.x >> 5;
  int lane = threadIdx.x & 31;

  // Preload the 16 A rows (16x128) as 4 K-chunk fragments (shared by all col tiles).
  bf16x16 a0 = load_a_frag(base + (size_t)row0 * D_ +  0, D_);
  bf16x16 a1 = load_a_frag(base + (size_t)row0 * D_ + 32, D_);
  bf16x16 a2 = load_a_frag(base + (size_t)row0 * D_ + 64, D_);
  bf16x16 a3 = load_a_frag(base + (size_t)row0 * D_ + 96, D_);

  for (int ct = wave; ct < NPER / 16; ct += 8) {
    int col0 = ct * 16;
    f32x8 c = {0.f, 0.f, 0.f, 0.f, 0.f, 0.f, 0.f, 0.f};
    c = wmma_bf16(a0, load_b_frag_T(base + (size_t)col0 * D_ +  0, D_), c);
    c = wmma_bf16(a1, load_b_frag_T(base + (size_t)col0 * D_ + 32, D_), c);
    c = wmma_bf16(a2, load_b_frag_T(base + (size_t)col0 * D_ + 64, D_), c);
    c = wmma_bf16(a3, load_b_frag_T(base + (size_t)col0 * D_ + 96, D_), c);
    int col = lane & 15;
    int rbase = (lane < 16) ? 0 : 8;
    float ncol = nrm[col0 + col];
#pragma unroll
    for (int r = 0; r < 8; ++r) {
      int row = rbase + r;
      float d2 = nrm[row0 + row] + ncol - 2.0f * c[r];
      if (row0 + row == col0 + col) d2 += 1.0e9f;  // mask self (matches eye*1e9)
      dist[row * NPER + col0 + col] = d2;
    }
  }
  __syncthreads();

  // Stage 1: 16 threads per row, each scans 64 candidates -> local top-16,
  // written back into this thread's own (disjoint) slice of the strip.
  // Outer loop kept rolled: the 16-deep insert body is already ~100 VALU ops.
  {
    int r   = threadIdx.x >> 4;
    int seg = threadIdx.x & 15;
    int cbase = seg * 64;
    float bd[KNN];
    int   bi[KNN];
#pragma unroll
    for (int k = 0; k < KNN; ++k) { bd[k] = 3.0e38f; bi[k] = 0; }
#pragma unroll 1
    for (int q = 0; q < 64; ++q) {
      int cI = cbase + q;
      TOPK_INSERT16(bd, bi, dist[r * NPER + cI], cI);
    }
#pragma unroll
    for (int k = 0; k < KNN; ++k) {
      dist[r * NPER + cbase + k]      = bd[k];
      dist[r * NPER + cbase + 16 + k] = __int_as_float(bi[k]);
    }
  }
  __syncthreads();

  // Stage 2: one thread per row merges 16 sorted 16-lists (early exit).
  // Both loops kept rolled to avoid a multi-KB straight-line code blob.
  if (threadIdx.x < 16) {
    int r = threadIdx.x;
    float bd[KNN];
    int   bi[KNN];
#pragma unroll
    for (int k = 0; k < KNN; ++k) { bd[k] = 3.0e38f; bi[k] = 0; }
#pragma unroll 1
    for (int s = 0; s < 16; ++s) {
      int lbase = r * NPER + s * 64;
#pragma unroll 1
      for (int k = 0; k < KNN; ++k) {
        float cd = dist[lbase + k];
        if (cd >= bd[KNN - 1]) break;  // source list sorted ascending
        int ci = __float_as_int(dist[lbase + 16 + k]);
        TOPK_INSERT16(bd, bi, cd, ci);
      }
    }
    int pt = b * NPER + row0 + r;
#pragma unroll
    for (int k = 0; k < KNN; ++k) idx_out[pt * KNN + k] = bi[k];
  }
}

// ---------------- EdgeConv: gather + [hi, hj-hi] MLP + max over neighbors --
// grid = NTOT/8 blocks of 256 threads; 8 points per block; weight fragments
// register-resident per wave (one 16-wide N-tile per wave).
__global__ __launch_bounds__(256) void edgeconv_kernel(
    const float* __restrict__ hb, const int* __restrict__ knn,
    const __bf16* __restrict__ w1, const float* __restrict__ b1,
    const __bf16* __restrict__ w2, const float* __restrict__ b2,
    float* __restrict__ hb_out) {
  __shared__ __bf16 e[16 * 256];   // 8 KB  edge features [neighbor][2D]
  __shared__ __bf16 m1[16 * 128];  // 4 KB  relu(e@W1+b1) in bf16
  int wave = threadIdx.x >> 5;
  int lane = threadIdx.x & 31;
  int col = lane & 15;
  int rbase = (lane < 16) ? 0 : 8;
  int n = wave * 16 + col;

  // Register-resident B fragments: 8 (layer1) + 4 (layer2) = 96 VGPRs/lane.
  bf16x16 wf1[8];
#pragma unroll
  for (int kc = 0; kc < 8; ++kc)
    wf1[kc] = load_b_frag(w1 + kc * 32 * D_ + wave * 16, D_);
  bf16x16 wf2[4];
#pragma unroll
  for (int kc = 0; kc < 4; ++kc)
    wf2[kc] = load_b_frag(w2 + kc * 32 * D_ + wave * 16, D_);
  float bias1 = b1[n];
  float bias2 = b2[n];

  int pt0 = blockIdx.x * 8;              // 8 consecutive points, same batch
  int bb  = pt0 >> 10;
  const float* hbase = hb + (size_t)bb * NPER * D_;
  int ej   = threadIdx.x >> 4;           // neighbor row this thread fills
  int eseg = threadIdx.x & 15;           // 16-col segment this thread fills

#pragma unroll 1
  for (int p = 0; p < 8; ++p) {
    int pt = pt0 + p;
    int local = pt & (NPER - 1);
    const float* hi = hbase + (size_t)local * D_;
    int nb = knn[pt * KNN + ej];
    const float* hj = hbase + (size_t)nb * D_;
#pragma unroll
    for (int q = 0; q < 16; ++q) {
      int c = eseg * 16 + q;
      float v = (c < D_) ? hi[c] : (hj[c - D_] - hi[c - D_]);
      e[ej * 256 + c] = (__bf16)v;
    }
    __syncthreads();

    // Layer 1: (16x256) @ (256x128), N-tile `wave`.
    f32x8 acc = {0.f, 0.f, 0.f, 0.f, 0.f, 0.f, 0.f, 0.f};
#pragma unroll
    for (int kc = 0; kc < 8; ++kc)
      acc = wmma_bf16(load_a_frag(&e[kc * 32], 256), wf1[kc], acc);
#pragma unroll
    for (int r = 0; r < 8; ++r)
      m1[(rbase + r) * D_ + n] = (__bf16)fmaxf(acc[r] + bias1, 0.f);
    __syncthreads();

    // Layer 2: (16x128) @ (128x128).
    f32x8 acc2 = {0.f, 0.f, 0.f, 0.f, 0.f, 0.f, 0.f, 0.f};
#pragma unroll
    for (int kc = 0; kc < 4; ++kc)
      acc2 = wmma_bf16(load_a_frag(&m1[kc * 32], D_), wf2[kc], acc2);

    // Max over the 16 neighbor rows: 8 rows in-register, other 8 in lane^16.
    float mx = acc2[0];
#pragma unroll
    for (int r = 1; r < 8; ++r) mx = fmaxf(mx, acc2[r]);
    mx = fmaxf(mx, __shfl_xor(mx, 16, 32));
    if (lane < 16) hb_out[(size_t)pt * D_ + n] = mx + bias2;
    // No extra barrier needed: next iteration's e-write is ordered behind the
    // post-layer1 barrier; next m1-write is ordered behind next post-e barrier.
  }
}

// ---------------- mean pool + 128->64->1 head ------------------------------
__global__ __launch_bounds__(128) void pool_head_kernel(
    const float* __restrict__ hb, const float* __restrict__ ow1,
    const float* __restrict__ ob1, const float* __restrict__ ow2,
    const float* __restrict__ ob2, float* __restrict__ out) {
  __shared__ float pooled[D_];
  __shared__ float h1[64];
  int b = blockIdx.x;
  int t = threadIdx.x;  // 128 threads, one per feature column
  const float* base = hb + (size_t)b * NPER * D_;
  float s = 0.f;
#pragma unroll 4
  for (int p = 0; p < NPER; ++p) s += base[(size_t)p * D_ + t];
  pooled[t] = s * (1.0f / (float)NPER);
  __syncthreads();
  if (t < 64) {
    float a = ob1[t];
#pragma unroll 8
    for (int c = 0; c < D_; ++c) a += pooled[c] * ow1[c * 64 + t];
    h1[t] = fmaxf(a, 0.f);
  }
  __syncthreads();
  if (t == 0) {
    float a = ob2[0];
#pragma unroll 8
    for (int k = 0; k < 64; ++k) a += h1[k] * ow2[k];
    out[b] = a;
  }
}

extern "C" void kernel_launch(void* const* d_in, const int* in_sizes, int n_in,
                              void* d_out, int out_size, void* d_ws, size_t ws_size,
                              hipStream_t stream) {
  (void)in_sizes; (void)n_in; (void)out_size; (void)ws_size;
  const float* x      = (const float*)d_in[0];
  // d_in[1] = pos (unused by reference), d_in[2] = batch (implicit via layout)
  const float* enc_w1 = (const float*)d_in[3];
  const float* enc_b1 = (const float*)d_in[4];
  const float* enc_w2 = (const float*)d_in[5];
  const float* enc_b2 = (const float*)d_in[6];
  const float* out_w1 = (const float*)d_in[7];
  const float* out_b1 = (const float*)d_in[8];
  const float* out_w2 = (const float*)d_in[9];
  const float* out_b2 = (const float*)d_in[10];
  const float* ec_w1[3] = {(const float*)d_in[11], (const float*)d_in[15], (const float*)d_in[19]};
  const float* ec_b1[3] = {(const float*)d_in[12], (const float*)d_in[16], (const float*)d_in[20]};
  const float* ec_w2[3] = {(const float*)d_in[13], (const float*)d_in[17], (const float*)d_in[21]};
  const float* ec_b2[3] = {(const float*)d_in[14], (const float*)d_in[18], (const float*)d_in[22]};

  // Workspace layout (~22.4 MB total)
  char* ws = (char*)d_ws;
  float*  hbA   = (float*)(ws);                         // 8 MB
  float*  hbB   = (float*)(ws + 8388608);               // 8 MB
  __bf16* hb16  = (__bf16*)(ws + 16777216);             // 4 MB
  float*  norms = (float*)(ws + 20971520);              // 64 KB
  int*    knn   = (int*)(ws + 21037056);                // 1 MB
  __bf16* w1b[3];
  __bf16* w2b[3];
  size_t off = 22085632;
  for (int l = 0; l < 3; ++l) { w1b[l] = (__bf16*)(ws + off); off += 2 * D_ * D_ * 2; }
  for (int l = 0; l < 3; ++l) { w2b[l] = (__bf16*)(ws + off); off += D_ * D_ * 2; }

  // Encoder
  encoder_kernel<<<NTOT / 256, 256, 0, stream>>>(x, enc_w1, enc_b1, enc_w2, enc_b2, hbA);

  // Weight conversions (fp32 -> bf16)
  for (int l = 0; l < 3; ++l) {
    cvt_bf16_kernel<<<(2 * D_ * D_ + 255) / 256, 256, 0, stream>>>(ec_w1[l], w1b[l], 2 * D_ * D_);
    cvt_bf16_kernel<<<(D_ * D_ + 255) / 256, 256, 0, stream>>>(ec_w2[l], w2b[l], D_ * D_);
  }

  // Three EdgeConv layers, ping-pong hbA <-> hbB
  float* cur = hbA;
  float* nxt = hbB;
  for (int l = 0; l < 3; ++l) {
    prep_kernel<<<NTOT / 8, 256, 0, stream>>>(cur, hb16, norms);
    knn_kernel<<<B_ * (NPER / 16), 256, 0, stream>>>(hb16, norms, knn);
    edgeconv_kernel<<<NTOT / 8, 256, 0, stream>>>(cur, knn, w1b[l], ec_b1[l], w2b[l], ec_b2[l], nxt);
    float* tmp = cur; cur = nxt; nxt = tmp;
  }

  // Mean pool + head -> (16,) fp32
  pool_head_kernel<<<B_, 128, 0, stream>>>(cur, out_w1, out_b1, out_w2, out_b2, (float*)d_out);
}
